// SemanticAlignmentModule_2336462209527
// MI455X (gfx1250) — matrix operand order repaired and checked by reading, not compiled, not run on hardware
//
#include <hip/hip_runtime.h>
#include <math.h>

using u16 = unsigned short;
typedef __attribute__((ext_vector_type(16))) __bf16 v16bf;
typedef __attribute__((ext_vector_type(8)))  float  v8f;

union FragBF { v16bf v; uint4 q[2]; };

__device__ __forceinline__ u16 f2bf(float f) {
    unsigned u = __float_as_uint(f);
    u += 0x7fffu + ((u >> 16) & 1u);
    return (u16)(u >> 16);
}

__device__ __forceinline__ float wsum(float x) {
#pragma unroll
    for (int m = 16; m; m >>= 1) x += __shfl_xor(x, m, 32);
    return x;
}
__device__ __forceinline__ float wmaxr(float x) {
#pragma unroll
    for (int m = 16; m; m >>= 1) x = fmaxf(x, __shfl_xor(x, m, 32));
    return x;
}
__device__ __forceinline__ v8f vzero() {
    v8f z;
#pragma unroll
    for (int i = 0; i < 8; i++) z[i] = 0.0f;
    return z;
}
__device__ __forceinline__ v8f wmma_bf(const FragBF& a, const FragBF& b, v8f c) {
    return __builtin_amdgcn_wmma_f32_16x16x32_bf16(false, a.v, false, b.v, (short)0, c, false, false);
}

// ---------------------------------------------------------------------------
// Convert fp32 weight (K,N) -> bf16 transposed (N,K)
// ---------------------------------------------------------------------------
__global__ void wconv_kernel(const float* __restrict__ W, u16* __restrict__ WT, int K, int N) {
    int idx = blockIdx.x * 256 + threadIdx.x;
    if (idx < K * N) {
        int k = idx / N, n = idx % N;
        WT[(size_t)n * K + k] = f2bf(W[idx]);
    }
}

// ---------------------------------------------------------------------------
// Text prep: bf16 convert + pad mask (sum|x| <= 1e-6). One wave per row (512).
// ---------------------------------------------------------------------------
__global__ void text_prep_kernel(const float* __restrict__ text, u16* __restrict__ tb,
                                 float* __restrict__ pad, int rows) {
    int wave = threadIdx.x >> 5, lane = threadIdx.x & 31;
    int row = blockIdx.x * 8 + wave;
    if (row >= rows) return;
    const float4* r4 = (const float4*)(text + (size_t)row * 512);
    float4 v[4];
    float as = 0.0f;
#pragma unroll
    for (int i = 0; i < 4; i++) {
        v[i] = r4[lane + 32 * i];
        as += fabsf(v[i].x) + fabsf(v[i].y) + fabsf(v[i].z) + fabsf(v[i].w);
    }
    as = wsum(as);
    if (lane == 0) pad[row] = (as <= 1e-6f) ? 1.0f : 0.0f;
    u16* tr = tb + (size_t)row * 512;
#pragma unroll
    for (int i = 0; i < 4; i++) {
        uint2 pk;
        pk.x = (unsigned)f2bf(v[i].x) | ((unsigned)f2bf(v[i].y) << 16);
        pk.y = (unsigned)f2bf(v[i].z) | ((unsigned)f2bf(v[i].w) << 16);
        ((uint2*)tr)[lane + 32 * i] = pk;
    }
}

// ---------------------------------------------------------------------------
// LayerNorm1 + gate dot. One wave per token (C=512).
// Writes x fp32, x bf16, gate = sigmoid(x@Wg + bg).
// ---------------------------------------------------------------------------
__global__ void ln_gate_kernel(const float* __restrict__ vis, const float* __restrict__ w,
                               const float* __restrict__ b, const float* __restrict__ Wg,
                               const float* __restrict__ bg, float* __restrict__ x,
                               u16* __restrict__ xb, float* __restrict__ gate) {
    int wave = threadIdx.x >> 5, lane = threadIdx.x & 31;
    int tok = blockIdx.x * 8 + wave;
    const float4* r4 = (const float4*)(vis + (size_t)tok * 512);
    float4 v[4];
    float s = 0.0f;
#pragma unroll
    for (int i = 0; i < 4; i++) { v[i] = r4[lane + 32 * i]; s += v[i].x + v[i].y + v[i].z + v[i].w; }
    float mu = wsum(s) * (1.0f / 512.0f);
    float vs = 0.0f;
#pragma unroll
    for (int i = 0; i < 4; i++) {
        float dx = v[i].x - mu, dy = v[i].y - mu, dz = v[i].z - mu, dw = v[i].w - mu;
        vs += dx * dx + dy * dy + dz * dz + dw * dw;
    }
    float rstd = rsqrtf(wsum(vs) * (1.0f / 512.0f) + 1e-5f);
    float4* xr = (float4*)(x + (size_t)tok * 512);
    u16* xbr = xb + (size_t)tok * 512;
    float gd = 0.0f;
#pragma unroll
    for (int i = 0; i < 4; i++) {
        int j = lane + 32 * i;
        float4 wv = ((const float4*)w)[j];
        float4 bv = ((const float4*)b)[j];
        float4 gv = ((const float4*)Wg)[j];
        float4 y;
        y.x = (v[i].x - mu) * rstd * wv.x + bv.x;
        y.y = (v[i].y - mu) * rstd * wv.y + bv.y;
        y.z = (v[i].z - mu) * rstd * wv.z + bv.z;
        y.w = (v[i].w - mu) * rstd * wv.w + bv.w;
        xr[j] = y;
        uint2 pk;
        pk.x = (unsigned)f2bf(y.x) | ((unsigned)f2bf(y.y) << 16);
        pk.y = (unsigned)f2bf(y.z) | ((unsigned)f2bf(y.w) << 16);
        ((uint2*)xbr)[j] = pk;
        gd += y.x * gv.x + y.y * gv.y + y.z * gv.z + y.w * gv.w;
    }
    gd = wsum(gd);
    if (lane == 0) gate[tok] = 1.0f / (1.0f + expf(-(gd + bg[0])));
}

// ---------------------------------------------------------------------------
// Row L2-normalize in place: row /= max(||row||, 1e-6). One wave per row.
// ---------------------------------------------------------------------------
__global__ void rownorm_kernel(float* __restrict__ buf, int rows) {
    int wave = threadIdx.x >> 5, lane = threadIdx.x & 31;
    int row = blockIdx.x * 8 + wave;
    if (row >= rows) return;
    float4* r4 = (float4*)(buf + (size_t)row * 512);
    float4 v[4];
    float ss = 0.0f;
#pragma unroll
    for (int i = 0; i < 4; i++) {
        v[i] = r4[lane + 32 * i];
        ss += v[i].x * v[i].x + v[i].y * v[i].y + v[i].z * v[i].z + v[i].w * v[i].w;
    }
    float sc = 1.0f / fmaxf(sqrtf(wsum(ss)), 1e-6f);
#pragma unroll
    for (int i = 0; i < 4; i++) {
        float4 y = v[i];
        y.x *= sc; y.y *= sc; y.z *= sc; y.w *= sc;
        r4[lane + 32 * i] = y;
    }
}

// ---------------------------------------------------------------------------
// Generic bf16 WMMA GEMM: OUT(M,N) = A(M,K)bf16 @ BT(N,K)bf16^T + bias
// EPI==1: multiply each output row by rowscale[row].
// Block = 256 thr (8 waves). Block tile 128(M) x 128(N); wave tile 16 x 128.
// ---------------------------------------------------------------------------
template <int EPI>
__global__ void gemm_bf16_kernel(const u16* __restrict__ A, const u16* __restrict__ BT,
                                 const float* __restrict__ bias, const float* __restrict__ rowscale,
                                 float* __restrict__ OUT, int M, int N, int K) {
    int wave = threadIdx.x >> 5, lane = threadIdx.x & 31;
    int half = lane >> 4, lm = lane & 15;
    int m = blockIdx.x * 128 + wave * 16 + lm;
    int mc = (m < M) ? m : (M - 1);
    const u16* Ap = A + (size_t)mc * K + 8 * half;
    const u16* Bp[8];
#pragma unroll
    for (int j = 0; j < 8; j++) {
        int n = blockIdx.y * 128 + j * 16 + lm;
        Bp[j] = BT + (size_t)n * K + 8 * half;
    }
    v8f acc[8];
#pragma unroll
    for (int j = 0; j < 8; j++) acc[j] = vzero();

    for (int k = 0; k < K; k += 32) {
        FragBF a;
        a.q[0] = *(const uint4*)(Ap + k);
        a.q[1] = *(const uint4*)(Ap + k + 16);
#pragma unroll
        for (int j = 0; j < 8; j++) {
            FragBF bb;
            bb.q[0] = *(const uint4*)(Bp[j] + k);
            bb.q[1] = *(const uint4*)(Bp[j] + k + 16);
            acc[j] = wmma_bf(a, bb, acc[j]);
        }
    }
    int mbase = blockIdx.x * 128 + wave * 16 + 8 * half;
#pragma unroll
    for (int j = 0; j < 8; j++) {
        int n = blockIdx.y * 128 + j * 16 + lm;
        float bv = bias[n];
#pragma unroll
        for (int r = 0; r < 8; r++) {
            int mm = mbase + r;
            if (mm < M) {
                float val = acc[j][r] + bv;
                if (EPI == 1) val *= rowscale[mm];
                OUT[(size_t)mm * N + n] = val;
            }
        }
    }
}

// ---------------------------------------------------------------------------
// Attention: one wave per (b,n) token, all 8 heads. Top-3 mask, softmax,
// entropy confidence -> bypass; writes aligned (bf16) for Wo GEMM.
// ---------------------------------------------------------------------------
__global__ void attn_kernel(const float* __restrict__ q, const float* __restrict__ kbuf,
                            const float* __restrict__ vbuf, const float* __restrict__ pad,
                            const float* __restrict__ ls_ptr, u16* __restrict__ alignedb,
                            float* __restrict__ bypass) {
    __shared__ float qs[8][512];
    __shared__ float at[8][96];
    int wave = threadIdx.x >> 5, lane = threadIdx.x & 31;
    int tok = blockIdx.x * 8 + wave;
    int b = tok >> 12;  // N = 4096 tokens per batch

    const float4* qr = (const float4*)(q + (size_t)tok * 512);
#pragma unroll
    for (int i = 0; i < 4; i++) ((float4*)qs[wave])[lane + 32 * i] = qr[lane + 32 * i];

    float ls = ls_ptr[0];
    ls = fminf(fmaxf(ls, -2.0f), 2.0f);
    float scale = expf(ls) * 0.125f;  // / sqrt(64)

    int tv[3];
    float padv[3];
#pragma unroll
    for (int i = 0; i < 3; i++) {
        tv[i] = lane + 32 * i;
        padv[i] = (tv[i] < 77) ? pad[b * 77 + tv[i]] : 1.0f;
    }
    float pc = 0.0f;
#pragma unroll
    for (int i = 0; i < 3; i++) if (tv[i] < 77) pc += padv[i];
    pc = wsum(pc);
    bool allpad = (pc >= 76.5f);

    float confacc = 0.0f;
    for (int h = 0; h < 8; h++) {
        float s[3];
#pragma unroll
        for (int i = 0; i < 3; i++) {
            if (tv[i] < 77 && padv[i] < 0.5f) {
                const float4* kr = (const float4*)(kbuf + ((size_t)(b * 77 + tv[i])) * 512 + h * 64);
                const float4* qh = (const float4*)(&qs[wave][h * 64]);
                float d = 0.0f;
#pragma unroll
                for (int dd = 0; dd < 16; dd++) {
                    float4 kv = kr[dd];
                    float4 qv = qh[dd];
                    d += kv.x * qv.x + kv.y * qv.y + kv.z * qv.z + kv.w * qv.w;
                }
                s[i] = d * scale;
            } else {
                s[i] = -INFINITY;
            }
        }
        // local sort desc
        float a0 = s[0], a1 = s[1], a2 = s[2], t;
        if (a0 < a1) { t = a0; a0 = a1; a1 = t; }
        if (a1 < a2) { t = a1; a1 = a2; a2 = t; }
        if (a0 < a1) { t = a0; a0 = a1; a1 = t; }
        // wave merge of sorted triples -> global top-3
#pragma unroll
        for (int m = 16; m; m >>= 1) {
            float b0 = __shfl_xor(a0, m, 32);
            float b1 = __shfl_xor(a1, m, 32);
            float b2 = __shfl_xor(a2, m, 32);
            if (b0 > a0) { t = a0; a0 = b0; b0 = t; t = a1; a1 = b1; b1 = t; t = a2; a2 = b2; b2 = t; }
            float m1 = fmaxf(a1, b0);
            float m2 = (a1 >= b0) ? fmaxf(a2, b0) : fmaxf(a1, b1);
            a1 = m1; a2 = m2;
        }
        float thr = a2, mx = a0;
        float e[3];
        float esum = 0.0f;
#pragma unroll
        for (int i = 0; i < 3; i++) {
            float ev = 0.0f;
            if (tv[i] < 77 && mx > -INFINITY && s[i] >= thr) ev = expf(s[i] - mx);
            e[i] = ev;
            esum += ev;
        }
        esum = wsum(esum);
        float inv = (esum > 0.0f && !allpad) ? 1.0f / esum : 0.0f;
        float lmax = 0.0f, lcnt = 0.0f, lH = 0.0f;
#pragma unroll
        for (int i = 0; i < 3; i++) {
            if (tv[i] < 77) {
                float p = e[i] * inv;
                at[wave][tv[i]] = p;
                lmax = fmaxf(lmax, p);
                lcnt += (p > 0.0f) ? 1.0f : 0.0f;
                float pcl = fmaxf(p, 1e-8f);
                lH += -pcl * logf(pcl);
            }
        }
        float mp = wmaxr(lmax);
        float tc = wsum(lcnt);
        float H = wsum(lH);
        float teff = fmaxf(tc, 2.0f);
        float ent = fmaxf(H / logf(teff), 0.0f);
        confacc += fminf(fmaxf(mp * (1.0f - ent), 0.0f), 1.0f);

        // aligned[h*64+d] = sum_t attn[t]*v[t][h*64+d]; 2 dims per lane
#pragma unroll
        for (int dp = 0; dp < 2; dp++) {
            int d = lane + 32 * dp;
            const float* vcol = vbuf + ((size_t)(b * 77)) * 512 + h * 64 + d;
            float acc = 0.0f;
            for (int tt = 0; tt < 77; tt++) acc += at[wave][tt] * vcol[(size_t)tt * 512];
            alignedb[(size_t)tok * 512 + h * 64 + d] = f2bf(acc);
        }
    }
    if (lane == 0) bypass[tok] = (confacc * 0.125f >= 0.35f) ? 1.0f : 0.0f;
}

// ---------------------------------------------------------------------------
// Combine: y = x + alpha*gate*proj (proj already has bias+bypass), then LN2.
// Writes y_ln fp32 (in place over proj) and y_ln bf16. One wave per token.
// ---------------------------------------------------------------------------
__global__ void combine_ln2_kernel(const float* __restrict__ x, const float* __restrict__ gate,
                                   const float* __restrict__ alpha_ptr, const float* __restrict__ w,
                                   const float* __restrict__ b, float* __restrict__ proj_yln,
                                   u16* __restrict__ ybf) {
    int wave = threadIdx.x >> 5, lane = threadIdx.x & 31;
    int tok = blockIdx.x * 8 + wave;
    float ag = alpha_ptr[0] * gate[tok];
    const float4* xr = (const float4*)(x + (size_t)tok * 512);
    float4* pr = (float4*)(proj_yln + (size_t)tok * 512);
    float4 v[4];
    float s = 0.0f;
#pragma unroll
    for (int i = 0; i < 4; i++) {
        float4 xv = xr[lane + 32 * i];
        float4 pv = pr[lane + 32 * i];
        float4 y;
        y.x = xv.x + ag * pv.x; y.y = xv.y + ag * pv.y;
        y.z = xv.z + ag * pv.z; y.w = xv.w + ag * pv.w;
        v[i] = y;
        s += y.x + y.y + y.z + y.w;
    }
    float mu = wsum(s) * (1.0f / 512.0f);
    float vs = 0.0f;
#pragma unroll
    for (int i = 0; i < 4; i++) {
        float dx = v[i].x - mu, dy = v[i].y - mu, dz = v[i].z - mu, dw = v[i].w - mu;
        vs += dx * dx + dy * dy + dz * dz + dw * dw;
    }
    float rstd = rsqrtf(wsum(vs) * (1.0f / 512.0f) + 1e-5f);
    u16* yr = ybf + (size_t)tok * 512;
#pragma unroll
    for (int i = 0; i < 4; i++) {
        int j = lane + 32 * i;
        float4 wv = ((const float4*)w)[j];
        float4 bv = ((const float4*)b)[j];
        float4 y;
        y.x = (v[i].x - mu) * rstd * wv.x + bv.x;
        y.y = (v[i].y - mu) * rstd * wv.y + bv.y;
        y.z = (v[i].z - mu) * rstd * wv.z + bv.z;
        y.w = (v[i].w - mu) * rstd * wv.w + bv.w;
        pr[j] = y;
        uint2 pk;
        pk.x = (unsigned)f2bf(y.x) | ((unsigned)f2bf(y.y) << 16);
        pk.y = (unsigned)f2bf(y.z) | ((unsigned)f2bf(y.w) << 16);
        ((uint2*)yr)[j] = pk;
    }
}

// ---------------------------------------------------------------------------
// Fused MLP: out = y_ln + gelu(y_ln@W1 + b1) @ W2 + b2.
// Block = 16 tokens. Hidden (2048) processed in two 1024-wide chunks staged
// in LDS as bf16; both GEMMs via WMMA. LDS = 16KB (y) + 32KB (t) = 48KB.
// ---------------------------------------------------------------------------
__global__ void mlp_kernel(const u16* __restrict__ ybf, const u16* __restrict__ W1T,
                           const float* __restrict__ b1, const u16* __restrict__ W2T,
                           const float* __restrict__ b2, const float* __restrict__ yln,
                           float* __restrict__ out) {
    __shared__ u16 yb_s[16 * 512];
    __shared__ u16 t_s[16 * 1024];
    int wave = threadIdx.x >> 5, lane = threadIdx.x & 31;
    int half = lane >> 4, lm = lane & 15;
    int tok0 = blockIdx.x * 16;

    // phase 1: stage 16x512 bf16 activations into LDS
    {
        const uint4* src = (const uint4*)(ybf + (size_t)tok0 * 512);
        uint4* dst = (uint4*)yb_s;
#pragma unroll
        for (int j = 0; j < 4; j++) {
            int idx = threadIdx.x + 256 * j;
            dst[idx] = src[idx];
        }
    }
    __syncthreads();

    v8f oacc[4];
#pragma unroll
    for (int j = 0; j < 4; j++) oacc[j] = vzero();

    for (int chunk = 0; chunk < 2; chunk++) {
        int hid0 = chunk * 1024;
        // phase 2: t[16 x 1024] = gelu(y @ W1[:, hid0:hid0+1024] + b1)
#pragma unroll
        for (int jt = 0; jt < 8; jt++) {
            int tile = wave * 8 + jt;
            int nloc = tile * 16 + lm;
            int ng = hid0 + nloc;
            const u16* Bp = W1T + (size_t)ng * 512 + 8 * half;
            const u16* Ap = yb_s + lm * 512 + 8 * half;
            v8f acc = vzero();
            for (int kk = 0; kk < 512; kk += 32) {
                FragBF a, bb;
                a.q[0] = *(const uint4*)(Ap + kk);
                a.q[1] = *(const uint4*)(Ap + kk + 16);
                bb.q[0] = *(const uint4*)(Bp + kk);
                bb.q[1] = *(const uint4*)(Bp + kk + 16);
                acc = wmma_bf(a, bb, acc);
            }
            float bias1 = b1[ng];
#pragma unroll
            for (int r = 0; r < 8; r++) {
                int ml = r + 8 * half;
                float v = acc[r] + bias1;
                v = 0.5f * v * (1.0f + erff(v * 0.70710678f));  // exact gelu
                t_s[ml * 1024 + nloc] = f2bf(v);
            }
        }
        __syncthreads();
        // phase 3: oacc += t @ W2[hid0:hid0+1024, :] ; wave handles 64 out cols
        const u16* Ap = t_s + lm * 1024 + 8 * half;
        for (int kk = 0; kk < 1024; kk += 32) {
            FragBF a;
            a.q[0] = *(const uint4*)(Ap + kk);
            a.q[1] = *(const uint4*)(Ap + kk + 16);
#pragma unroll
            for (int j = 0; j < 4; j++) {
                int n = wave * 64 + j * 16 + lm;
                const u16* Bp = W2T + (size_t)n * 2048 + hid0 + 8 * half;
                FragBF bb;
                bb.q[0] = *(const uint4*)(Bp + kk);
                bb.q[1] = *(const uint4*)(Bp + kk + 16);
                oacc[j] = wmma_bf(a, bb, oacc[j]);
            }
        }
        __syncthreads();
    }
    // epilogue: + b2 + residual y_ln
#pragma unroll
    for (int j = 0; j < 4; j++) {
        int n = wave * 64 + j * 16 + lm;
        float bv = b2[n];
#pragma unroll
        for (int r = 0; r < 8; r++) {
            int ml = r + 8 * half;
            size_t idx = (size_t)(tok0 + ml) * 512 + n;
            out[idx] = oacc[j][r] + bv + yln[idx];
        }
    }
}

// ---------------------------------------------------------------------------
// Launch
// ---------------------------------------------------------------------------
extern "C" void kernel_launch(void* const* d_in, const int* in_sizes, int n_in,
                              void* d_out, int out_size, void* d_ws, size_t ws_size,
                              hipStream_t stream) {
    const int TOK = 32768;      // B*H*W = 8*64*64
    const int C = 512;
    const int TT = 616;         // B*T = 8*77

    const float* vis   = (const float*)d_in[0];
    const float* text  = (const float*)d_in[1];
    const float* n1w   = (const float*)d_in[2];
    const float* n1b   = (const float*)d_in[3];
    const float* n2w   = (const float*)d_in[4];
    const float* n2b   = (const float*)d_in[5];
    const float* Wq    = (const float*)d_in[6];
    const float* bq    = (const float*)d_in[7];
    const float* Wk    = (const float*)d_in[8];
    const float* bk    = (const float*)d_in[9];
    const float* Wv    = (const float*)d_in[10];
    const float* bv    = (const float*)d_in[11];
    const float* Wo    = (const float*)d_in[12];
    const float* bo    = (const float*)d_in[13];
    const float* Wg    = (const float*)d_in[14];
    const float* bg    = (const float*)d_in[15];
    const float* lscal = (const float*)d_in[16];
    const float* W1    = (const float*)d_in[17];
    const float* b1    = (const float*)d_in[18];
    const float* W2    = (const float*)d_in[19];
    const float* b2    = (const float*)d_in[20];
    const float* alpha = (const float*)d_in[21];

    char* w = (char*)d_ws;
    auto al = [](size_t v) { return (v + 255) & ~(size_t)255; };
    size_t o = 0;
    size_t o_x   = o; o = al(o + (size_t)TOK * C * 4);   // x fp32
    size_t o_xb  = o; o = al(o + (size_t)TOK * C * 2);   // x bf16, later aligned bf16
    size_t o_q   = o; o = al(o + (size_t)TOK * C * 4);   // q fp32 -> proj -> y_ln
    size_t o_ybf = o; o = al(o + (size_t)TOK * C * 2);   // y_ln bf16
    size_t o_gt  = o; o = al(o + (size_t)TOK * 4);
    size_t o_by  = o; o = al(o + (size_t)TOK * 4);
    size_t o_pd  = o; o = al(o + (size_t)TT * 4);
    size_t o_tb  = o; o = al(o + (size_t)TT * C * 2);
    size_t o_k   = o; o = al(o + (size_t)TT * C * 4);
    size_t o_v   = o; o = al(o + (size_t)TT * C * 4);
    size_t o_WqT = o; o = al(o + (size_t)C * C * 2);
    size_t o_WkT = o; o = al(o + (size_t)C * C * 2);
    size_t o_WvT = o; o = al(o + (size_t)C * C * 2);
    size_t o_WoT = o; o = al(o + (size_t)C * C * 2);
    size_t o_W1T = o; o = al(o + (size_t)C * 2048 * 2);
    size_t o_W2T = o; o = al(o + (size_t)2048 * C * 2);
    (void)ws_size; (void)n_in; (void)in_sizes; (void)out_size;

    float* xf   = (float*)(w + o_x);
    u16*   xb   = (u16*)(w + o_xb);
    float* qb   = (float*)(w + o_q);
    u16*   ybf  = (u16*)(w + o_ybf);
    float* gate = (float*)(w + o_gt);
    float* byp  = (float*)(w + o_by);
    float* pad  = (float*)(w + o_pd);
    u16*   tb   = (u16*)(w + o_tb);
    float* kb   = (float*)(w + o_k);
    float* vb   = (float*)(w + o_v);
    u16*   WqT  = (u16*)(w + o_WqT);
    u16*   WkT  = (u16*)(w + o_WkT);
    u16*   WvT  = (u16*)(w + o_WvT);
    u16*   WoT  = (u16*)(w + o_WoT);
    u16*   W1T  = (u16*)(w + o_W1T);
    u16*   W2T  = (u16*)(w + o_W2T);

    // weight conversion / transposition
    int nqk = C * C;
    wconv_kernel<<<(nqk + 255) / 256, 256, 0, stream>>>(Wq, WqT, C, C);
    wconv_kernel<<<(nqk + 255) / 256, 256, 0, stream>>>(Wk, WkT, C, C);
    wconv_kernel<<<(nqk + 255) / 256, 256, 0, stream>>>(Wv, WvT, C, C);
    wconv_kernel<<<(nqk + 255) / 256, 256, 0, stream>>>(Wo, WoT, C, C);
    int n1 = C * 2048;
    wconv_kernel<<<(n1 + 255) / 256, 256, 0, stream>>>(W1, W1T, C, 2048);
    wconv_kernel<<<(n1 + 255) / 256, 256, 0, stream>>>(W2, W2T, 2048, C);

    text_prep_kernel<<<77, 256, 0, stream>>>(text, tb, pad, TT);
    ln_gate_kernel<<<TOK / 8, 256, 0, stream>>>(vis, n1w, n1b, Wg, bg, xf, xb, gate);

    // projections
    gemm_bf16_kernel<0><<<dim3(TOK / 128, C / 128), 256, 0, stream>>>(xb, WqT, bq, nullptr, qb, TOK, C, C);
    gemm_bf16_kernel<0><<<dim3((TT + 127) / 128, C / 128), 256, 0, stream>>>(tb, WkT, bk, nullptr, kb, TT, C, C);
    gemm_bf16_kernel<0><<<dim3((TT + 127) / 128, C / 128), 256, 0, stream>>>(tb, WvT, bv, nullptr, vb, TT, C, C);

    rownorm_kernel<<<TOK / 8, 256, 0, stream>>>(qb, TOK);
    rownorm_kernel<<<77, 256, 0, stream>>>(kb, TT);

    // attention (VALU): writes aligned bf16 into xb, bypass per token
    attn_kernel<<<TOK / 8, 256, 0, stream>>>(qb, kb, vb, pad, lscal, xb, byp);

    // output projection with bypass row-scale; writes into qb (q is dead)
    gemm_bf16_kernel<1><<<dim3(TOK / 128, C / 128), 256, 0, stream>>>(xb, WoT, bo, byp, qb, TOK, C, C);

    // residual + LN2 (in place over qb), bf16 copy for MLP
    combine_ln2_kernel<<<TOK / 8, 256, 0, stream>>>(xf, gate, alpha, n2w, n2b, qb, ybf);

    // fused MLP with residual -> d_out
    mlp_kernel<<<TOK / 16, 256, 0, stream>>>(ybf, W1T, b1, W2T, b2, qb, (float*)d_out);
}
